// ImprovedLSTMAttention_40398462386778
// MI455X (gfx1250) — matrix operand order repaired and checked by reading, not compiled, not run on hardware
//
#include <hip/hip_runtime.h>
#include <hip/hip_bf16.h>

typedef __attribute__((ext_vector_type(16))) _Float16 v16h;
typedef __attribute__((ext_vector_type(8)))  _Float16 v8h;
typedef __attribute__((ext_vector_type(8)))  float    v8f;

#define BSZ   8
#define TLEN  256
#define EDIM  128
#define HDIM  256
#define VOC   32000
#define MROWS (BSZ * TLEN)   // 2048

__device__ __forceinline__ v16h make_afrag(const _Float16* p) {
  v8h a0 = *(const v8h*)(p);        // K = k + hi*8      .. +7
  v8h a1 = *(const v8h*)(p + 16);   // K = k + 16 + hi*8 .. +7
  return __builtin_shufflevector(a0, a1, 0,1,2,3,4,5,6,7,8,9,10,11,12,13,14,15);
}

__device__ __forceinline__ void store_tile(float* __restrict__ C, int N,
                                           int m0, int n, int hi,
                                           const v8f& acc, float bn, int act, int nt) {
  #pragma unroll
  for (int v = 0; v < 8; v++) {
    int m = m0 + v + hi * 8;                 // C/D layout: VGPR v, half hi
    float val = acc[v] + bn;
    if (act == 1) val = tanhf(val);
    float* dst = C + (size_t)m * N + n;
    if (nt) __builtin_nontemporal_store(val, dst);
    else    *dst = val;
  }
}

// ---------------------------------------------------------------------------
// Embedding gather -> f16 A-matrix [B*T, E]
// ---------------------------------------------------------------------------
__global__ __launch_bounds__(256) void embed_kernel(const int* __restrict__ x,
                                                    const float* __restrict__ emb,
                                                    _Float16* __restrict__ out) {
  int idx = blockIdx.x * blockDim.x + threadIdx.x;   // over 2048*128
  if (idx >= MROWS * EDIM) return;
  int row = idx >> 7;           // b*T + t
  int col = idx & (EDIM - 1);
  int tok = x[row];
  out[idx] = (_Float16)emb[(size_t)tok * EDIM + col];
}

// ---------------------------------------------------------------------------
// f32 -> f16 weight conversion
// ---------------------------------------------------------------------------
__global__ __launch_bounds__(256) void cvt_f16_kernel(const float* __restrict__ in,
                                                      _Float16* __restrict__ out, int n) {
  int i = blockIdx.x * blockDim.x + threadIdx.x;
  if (i < n) out[i] = (_Float16)in[i];
}

// ---------------------------------------------------------------------------
// WMMA GEMM, 2x2 register blocking: each wave owns a 32(M) x 32(N) tile.
// C[M,N] = act( A[M,K](f16) @ Bt[N,K](f16)^T + bias[N] )
// grid: (N/256, M/32), block 256 (8 waves); wave w owns n0 = bx*256 + w*32.
// Per K-step: 4 A-loads + 4 B-loads feed 4 WMMAs (vs 4 loads / 1 WMMA before).
// ---------------------------------------------------------------------------
__global__ __launch_bounds__(256) void wmma_gemm_kernel(
    const _Float16* __restrict__ A, const _Float16* __restrict__ Bt,
    const float* __restrict__ bias, float* __restrict__ C,
    int M, int N, int K, int act, int nt) {
  const int lane = threadIdx.x & 31;
  const int wave = threadIdx.x >> 5;
  const int lm = lane & 15;
  const int hi = lane >> 4;
  const int m0 = blockIdx.y * 32;
  const int n0 = blockIdx.x * 256 + wave * 32;

  const _Float16* arow0 = A + (size_t)(m0 + lm) * K + hi * 8;
  const _Float16* arow1 = arow0 + (size_t)16 * K;
  const _Float16* brow0 = Bt + (size_t)(n0 + lm) * K + hi * 8;
  const _Float16* brow1 = brow0 + (size_t)16 * K;

  v8f acc00 = {}, acc01 = {}, acc10 = {}, acc11 = {};
  for (int k = 0; k < K; k += 32) {
    __builtin_prefetch(brow0 + k + 256, 0, 3);            // global_prefetch_b8
    __builtin_prefetch(brow1 + k + 256, 0, 3);
    v16h a0 = make_afrag(arow0 + k);
    v16h a1 = make_afrag(arow1 + k);
    v16h b0 = make_afrag(brow0 + k);                      // K = k+hi*8..+7, k+16+hi*8..+7
    v16h b1 = make_afrag(brow1 + k);
    acc00 = __builtin_amdgcn_wmma_f32_16x16x32_f16(false, a0, false, b0, (short)0, acc00, false, false);
    acc01 = __builtin_amdgcn_wmma_f32_16x16x32_f16(false, a0, false, b1, (short)0, acc01, false, false);
    acc10 = __builtin_amdgcn_wmma_f32_16x16x32_f16(false, a1, false, b0, (short)0, acc10, false, false);
    acc11 = __builtin_amdgcn_wmma_f32_16x16x32_f16(false, a1, false, b1, (short)0, acc11, false, false);
  }
  const float bn0 = bias[n0 + lm];
  const float bn1 = bias[n0 + 16 + lm];
  store_tile(C, N, m0,      n0 + lm,      hi, acc00, bn0, act, nt);
  store_tile(C, N, m0,      n0 + 16 + lm, hi, acc01, bn1, act, nt);
  store_tile(C, N, m0 + 16, n0 + lm,      hi, acc10, bn0, act, nt);
  store_tile(C, N, m0 + 16, n0 + 16 + lm, hi, acc11, bn1, act, nt);
}

// ---------------------------------------------------------------------------
// Persistent bidirectional LSTM recurrence, one workgroup per direction.
// grid = 2 (dir = blockIdx.x), block = 512 (16 waves).
// The h A-fragments (invariant across a wave's 4 gate-column tiles) are
// hoisted into registers once per timestep: 16 ds loads instead of 64.
// ---------------------------------------------------------------------------
__global__ __launch_bounds__(512) void lstm_pair_kernel(
    const float* __restrict__ pre_f, const float* __restrict__ pre_b,
    const _Float16* __restrict__ Uf, const _Float16* __restrict__ Ub,
    float* __restrict__ hout32, _Float16* __restrict__ hout16,
    float* __restrict__ hT_base, float* __restrict__ cT_base) {
  __shared__ _Float16 h_lds[16][264];      // f16 A-matrix, rows 8..15 stay zero
  __shared__ float    c_lds[BSZ][HDIM];
  __shared__ float    g_lds[BSZ][4 * HDIM];

  const int dir = blockIdx.x;
  const float*     pre = dir ? pre_b : pre_f;
  const _Float16*  U   = dir ? Ub    : Uf;
  const int col0 = dir * HDIM;
  float* hT = hT_base + (size_t)dir * BSZ * HDIM;
  float* cT = cT_base + (size_t)dir * BSZ * HDIM;

  for (int i = threadIdx.x; i < 16 * 264; i += blockDim.x) (&h_lds[0][0])[i] = (_Float16)0.f;
  for (int i = threadIdx.x; i < BSZ * HDIM; i += blockDim.x) (&c_lds[0][0])[i] = 0.f;
  __syncthreads();

  const int lane = threadIdx.x & 31;
  const int wave = threadIdx.x >> 5;       // 0..15
  const int lm = lane & 15;
  const int hi = lane >> 4;
  const _Float16* arow = &h_lds[lm][hi * 8];

  for (int t = 0; t < TLEN; t++) {
    const int tt = dir ? (TLEN - 1 - t) : t;
    // ---- hoisted A fragments for this timestep (h is tile-invariant) ----
    v16h afrag[HDIM / 32];
    #pragma unroll
    for (int kk = 0; kk < HDIM / 32; kk++) afrag[kk] = make_afrag(arow + kk * 32);
    // ---- phase 1: g[0:8, 0:1024] = h @ U^T + pre[tt] ----
    #pragma unroll
    for (int ntile = 0; ntile < 4; ntile++) {
      const int n0 = (wave * 4 + ntile) * 16;
      const _Float16* brow = U + (size_t)(n0 + lm) * HDIM + hi * 8;
      v8f acc = {};
      #pragma unroll
      for (int kk = 0; kk < HDIM / 32; kk++) {
        v16h b = make_afrag(brow + kk * 32);
        acc = __builtin_amdgcn_wmma_f32_16x16x32_f16(false, afrag[kk], false, b,
                                                     (short)0, acc, false, false);
      }
      if (hi == 0) {                       // lanes 0..15 hold M rows 0..7 in acc[0..7]
        #pragma unroll
        for (int v = 0; v < BSZ; v++)
          g_lds[v][n0 + lm] = acc[v] + pre[((size_t)v * TLEN + tt) * (4 * HDIM) + n0 + lm];
      }
    }
    __syncthreads();
    // ---- phase 2: gates, state update ----
    for (int e = threadIdx.x; e < BSZ * HDIM; e += blockDim.x) {
      const int b = e >> 8;
      const int n = e & (HDIM - 1);
      float ig = g_lds[b][n];
      float fg = g_lds[b][HDIM + n];
      float gg = g_lds[b][2 * HDIM + n];
      float og = g_lds[b][3 * HDIM + n];
      ig = 1.f / (1.f + __expf(-ig));
      fg = 1.f / (1.f + __expf(-fg));
      gg = tanhf(gg);
      og = 1.f / (1.f + __expf(-og));
      const float c = fg * c_lds[b][n] + ig * gg;
      c_lds[b][n] = c;
      const float h = og * tanhf(c);
      h_lds[b][n] = (_Float16)h;
      const size_t row = (size_t)b * TLEN + tt;
      hout16[row * 512 + col0 + n] = (_Float16)h;
      if (hout32) hout32[row * 512 + col0 + n] = h;
      if (t == TLEN - 1) { hT[b * HDIM + n] = h; cT[b * HDIM + n] = c; }
    }
    __syncthreads();
  }
}

// ---------------------------------------------------------------------------
// scores[r] = a[r,:] . aw2 + ab2       (r = b*T+t, K = 256)
// ---------------------------------------------------------------------------
__global__ __launch_bounds__(256) void scores_kernel(const float* __restrict__ a,
                                                     const float* __restrict__ aw2,
                                                     const float* __restrict__ ab2,
                                                     float* __restrict__ scores) {
  int r = blockIdx.x * blockDim.x + threadIdx.x;
  if (r >= MROWS) return;
  float acc = ab2[0];
  for (int k = 0; k < HDIM; k++) acc += a[(size_t)r * HDIM + k] * aw2[k];
  scores[r] = acc;
}

// ---------------------------------------------------------------------------
// Softmax over T, context = sum_t w*lstm_out, combined = lstm_out + context.
// grid = B (8), block = T (256).
// ---------------------------------------------------------------------------
__global__ __launch_bounds__(256) void attn_kernel(const float* __restrict__ scores,
                                                   const float* __restrict__ lo,
                                                   _Float16* __restrict__ comb) {
  __shared__ float w[TLEN];
  __shared__ float red[TLEN];
  __shared__ float ctx[512];
  const int b = blockIdx.x, tid = threadIdx.x;
  const float s = scores[b * TLEN + tid];
  red[tid] = s; __syncthreads();
  for (int o = 128; o > 0; o >>= 1) { if (tid < o) red[tid] = fmaxf(red[tid], red[tid + o]); __syncthreads(); }
  const float mx = red[0]; __syncthreads();
  const float e = __expf(s - mx);
  red[tid] = e; __syncthreads();
  for (int o = 128; o > 0; o >>= 1) { if (tid < o) red[tid] += red[tid + o]; __syncthreads(); }
  w[tid] = e / red[0]; __syncthreads();
  for (int n = tid; n < 512; n += 256) {
    float acc = 0.f;
    for (int t = 0; t < TLEN; t++) acc += w[t] * lo[((size_t)b * TLEN + t) * 512 + n];
    ctx[n] = acc;
  }
  __syncthreads();
  for (int t = 0; t < TLEN; t++)
    for (int n = tid; n < 512; n += 256) {
      const size_t idx = ((size_t)b * TLEN + t) * 512 + n;
      comb[idx] = (_Float16)(lo[idx] + ctx[n]);
    }
}

// ---------------------------------------------------------------------------
// LayerNorm + ReLU, one row (H=256) per block.
// ---------------------------------------------------------------------------
__global__ __launch_bounds__(256) void ln_relu_kernel(const float* __restrict__ h,
                                                      const float* __restrict__ g,
                                                      const float* __restrict__ bb,
                                                      _Float16* __restrict__ out) {
  __shared__ float red[HDIM];
  const int r = blockIdx.x, tid = threadIdx.x;
  const float v = h[(size_t)r * HDIM + tid];
  red[tid] = v; __syncthreads();
  for (int o = 128; o > 0; o >>= 1) { if (tid < o) red[tid] += red[tid + o]; __syncthreads(); }
  const float mu = red[0] * (1.f / HDIM); __syncthreads();
  const float d = v - mu;
  red[tid] = d * d; __syncthreads();
  for (int o = 128; o > 0; o >>= 1) { if (tid < o) red[tid] += red[tid + o]; __syncthreads(); }
  const float var = red[0] * (1.f / HDIM);
  float y = d * rsqrtf(var + 1e-5f) * g[tid] + bb[tid];
  y = fmaxf(y, 0.f);
  out[(size_t)r * HDIM + tid] = (_Float16)y;
}

// ---------------------------------------------------------------------------
extern "C" void kernel_launch(void* const* d_in, const int* in_sizes, int n_in,
                              void* d_out, int out_size, void* d_ws, size_t ws_size,
                              hipStream_t stream) {
  const int*   x      = (const int*)  d_in[0];
  const float* emb    = (const float*)d_in[1];
  const float* wih[4] = {(const float*)d_in[2], (const float*)d_in[5],
                         (const float*)d_in[8], (const float*)d_in[11]};
  const float* whh[4] = {(const float*)d_in[3], (const float*)d_in[6],
                         (const float*)d_in[9], (const float*)d_in[12]};
  const float* bia[4] = {(const float*)d_in[4], (const float*)d_in[7],
                         (const float*)d_in[10], (const float*)d_in[13]};
  const float* aw1 = (const float*)d_in[14];
  const float* ab1 = (const float*)d_in[15];
  const float* aw2 = (const float*)d_in[16];
  const float* ab2 = (const float*)d_in[17];
  const float* fc1w = (const float*)d_in[18];
  const float* fc1b = (const float*)d_in[19];
  const float* lng  = (const float*)d_in[20];
  const float* lnb  = (const float*)d_in[21];
  const float* fc2w = (const float*)d_in[22];
  const float* fc2b = (const float*)d_in[23];

  char* ws = (char*)d_ws;
  size_t o = 0;
  auto alloc = [&](size_t bytes) { size_t r = o; o += (bytes + 255) & ~(size_t)255; return r; };

  size_t o_emb16   = alloc((size_t)MROWS * EDIM * 2);
  size_t o_wih16[4], o_whh16[4], o_pre[4];
  const size_t wih_n[4] = {1024u * 128, 1024u * 128, 1024u * 512, 1024u * 512};
  for (int i = 0; i < 4; i++) o_wih16[i] = alloc(wih_n[i] * 2);
  for (int i = 0; i < 4; i++) o_whh16[i] = alloc((size_t)1024 * 256 * 2);
  for (int i = 0; i < 4; i++) o_pre[i]   = alloc((size_t)MROWS * 1024 * 4);
  size_t o_aw1_16  = alloc((size_t)256 * 512 * 2);
  size_t o_fc1w16  = alloc((size_t)256 * 512 * 2);
  size_t o_fc2w16  = alloc((size_t)VOC * 256 * 2);
  size_t o_y016    = alloc((size_t)MROWS * 512 * 2);
  size_t o_lo32    = alloc((size_t)MROWS * 512 * 4);
  size_t o_lo16    = alloc((size_t)MROWS * 512 * 2);
  size_t o_a32     = alloc((size_t)MROWS * 256 * 4);
  size_t o_scores  = alloc((size_t)MROWS * 4);
  size_t o_comb16  = alloc((size_t)MROWS * 512 * 2);
  size_t o_h32     = alloc((size_t)MROWS * 256 * 4);
  size_t o_hr16    = alloc((size_t)MROWS * 256 * 2);

  float* out = (float*)d_out;
  float* hn  = out + (size_t)MROWS * VOC;   // h_n [4,B,H]
  float* cn  = hn + 4 * BSZ * HDIM;         // c_n [4,B,H]

  // 1. embedding gather (f16)
  embed_kernel<<<(MROWS * EDIM + 255) / 256, 256, 0, stream>>>(
      x, emb, (_Float16*)(ws + o_emb16));

  // 2. weight conversions to f16
  auto cvt = [&](const float* src, size_t dstoff, int n) {
    cvt_f16_kernel<<<(n + 255) / 256, 256, 0, stream>>>(src, (_Float16*)(ws + dstoff), n);
  };
  for (int i = 0; i < 4; i++) cvt(wih[i], o_wih16[i], (int)wih_n[i]);
  for (int i = 0; i < 4; i++) cvt(whh[i], o_whh16[i], 1024 * 256);
  cvt(aw1, o_aw1_16, 256 * 512);
  cvt(fc1w, o_fc1w16, 256 * 512);
  cvt(fc2w, o_fc2w16, VOC * 256);

  auto gemm = [&](size_t aoff, size_t boff, const float* bias, float* C,
                  int N, int K, int act, int nt) {
    dim3 grid(N / 256, MROWS / 32);
    wmma_gemm_kernel<<<grid, 256, 0, stream>>>(
        (const _Float16*)(ws + aoff), (const _Float16*)(ws + boff),
        bias, C, MROWS, N, K, act, nt);
  };

  // 3. layer-0 pre-activations (bias folded here)
  gemm(o_emb16, o_wih16[0], bia[0], (float*)(ws + o_pre[0]), 1024, EDIM, 0, 0);
  gemm(o_emb16, o_wih16[1], bia[1], (float*)(ws + o_pre[1]), 1024, EDIM, 0, 0);

  // 4. layer-0 recurrence (fwd+bwd concurrently) -> y0 f16 [2048,512]
  lstm_pair_kernel<<<2, 512, 0, stream>>>(
      (const float*)(ws + o_pre[0]), (const float*)(ws + o_pre[1]),
      (const _Float16*)(ws + o_whh16[0]), (const _Float16*)(ws + o_whh16[1]),
      nullptr, (_Float16*)(ws + o_y016), hn + 0, cn + 0);

  // 5. layer-1 pre-activations (K = 512)
  gemm(o_y016, o_wih16[2], bia[2], (float*)(ws + o_pre[2]), 1024, 512, 0, 0);
  gemm(o_y016, o_wih16[3], bia[3], (float*)(ws + o_pre[3]), 1024, 512, 0, 0);

  // 6. layer-1 recurrence -> lstm_out f32+f16, h_n/c_n slots 2,3
  lstm_pair_kernel<<<2, 512, 0, stream>>>(
      (const float*)(ws + o_pre[2]), (const float*)(ws + o_pre[3]),
      (const _Float16*)(ws + o_whh16[2]), (const _Float16*)(ws + o_whh16[3]),
      (float*)(ws + o_lo32), (_Float16*)(ws + o_lo16),
      hn + 2 * BSZ * HDIM, cn + 2 * BSZ * HDIM);

  // 7. attention: a = tanh(lstm_out @ aw1^T + ab1)
  gemm(o_lo16, o_aw1_16, ab1, (float*)(ws + o_a32), 256, 512, 1, 0);

  // 8. scores = a @ aw2^T + ab2
  scores_kernel<<<MROWS / 256, 256, 0, stream>>>(
      (const float*)(ws + o_a32), aw2, ab2, (float*)(ws + o_scores));

  // 9. softmax + context + combined (f16)
  attn_kernel<<<BSZ, TLEN, 0, stream>>>(
      (const float*)(ws + o_scores), (const float*)(ws + o_lo32),
      (_Float16*)(ws + o_comb16));

  // 10. fc1
  gemm(o_comb16, o_fc1w16, fc1b, (float*)(ws + o_h32), 256, 512, 0, 0);

  // 11. LayerNorm + ReLU -> f16
  ln_relu_kernel<<<MROWS, HDIM, 0, stream>>>(
      (const float*)(ws + o_h32), lng, lnb, (_Float16*)(ws + o_hr16));

  // 12. fc2 -> d_out (262 MB, non-temporal stores)
  gemm(o_hr16, o_fc2w16, fc2b, out, VOC, 256, 0, 1);
}
